// attn_695784702709
// MI455X (gfx1250) — compile-verified, matrix operand order checked by reading, not run on hardware
//
#include <hip/hip_runtime.h>

// ---------------------------------------------------------------------------
// Causal MHA block for MI455X (gfx1250, wave32, WMMA bf16 16x16x32, f32 acc)
//   B=4, T=2048, C=1024, H=16, D=64
// Pipeline (all on `stream`):
//   1) convert x -> bf16 (vectorized), W_qkv/W_proj -> bf16^T (LDS transpose)
//   2) qkv GEMM (WMMA, 32x64/wave, reg double-buffered) + bias,
//      scatter to Q[B,H,T,D], K[B,H,T,D], Vt[B,H,D,T]
//   3) flash attention (WMMA scores + WMMA P*V, online softmax, LDS P-transpose,
//      V loads overlapped with softmax) -> attn bf16 [B,T,C]
//   4) proj GEMM (WMMA, 32x64/wave, reg double-buffered) + bias -> f32 [B,T,C]
// ---------------------------------------------------------------------------

typedef __attribute__((ext_vector_type(16))) __bf16 v16bf;
typedef __attribute__((ext_vector_type(8)))  __bf16 bf16x8;
typedef __attribute__((ext_vector_type(4)))  __bf16 bf16x4;
typedef __attribute__((ext_vector_type(8)))  float  v8f;
typedef __attribute__((ext_vector_type(4)))  float  f32x4;

#define TT   2048
#define CC   1024
#define HH   16
#define DD   64
#define BB   4
#define MROWS (BB * TT)          /* 8192 */
#define NQKV  (3 * CC)           /* 3072 */

static __device__ __forceinline__ __bf16 f2bf(float f) {
  union { float f; unsigned u; } a; a.f = f;
  unsigned u = a.u + 0x7FFFu + ((a.u >> 16) & 1u);   // round-to-nearest-even
  union { unsigned short s; __bf16 b; } o; o.s = (unsigned short)(u >> 16);
  return o.b;
}

// A/B fragment load for V_WMMA_*_16x16x32_bf16, per CDNA5 ISA 7.12.2:
// lane L: row = L&15, half = L>>4; elements are two contiguous 8x-bf16 (16B)
// chunks of that row at k = k0 + half*8 and k0 + 16 + half*8.
static __device__ __forceinline__ v16bf load_frag(const __bf16* base, int ld, int k0) {
  const int lane = threadIdx.x & 31;
  const int m  = lane & 15;
  const int hl = lane >> 4;
  const __bf16* p = base + (size_t)m * ld + k0 + hl * 8;
  bf16x8 lo = *(const bf16x8*)(p);
  bf16x8 hi = *(const bf16x8*)(p + 16);
  v16bf r;
#pragma unroll
  for (int i = 0; i < 8; ++i) { r[i] = lo[i]; r[i + 8] = hi[i]; }
  return r;
}

static __device__ __forceinline__ void prefetch_frag(const __bf16* base, int ld, int k0) {
  const int lane = threadIdx.x & 31;
  const int m  = lane & 15;
  const int hl = lane >> 4;
  __builtin_prefetch(base + (size_t)m * ld + k0 + hl * 8, 0, 1);  // global_prefetch_b8
}

// ---------------------------------------------------------------------------
// conversion kernels
// ---------------------------------------------------------------------------
__global__ __launch_bounds__(256) void cvt_f32_bf16_v4(const float* __restrict__ in,
                                                       __bf16* __restrict__ out, int n4) {
  int i = blockIdx.x * blockDim.x + threadIdx.x;
  if (i < n4) {
    f32x4 v = *(const f32x4*)(in + (size_t)i * 4);
    bf16x4 o;
#pragma unroll
    for (int c = 0; c < 4; ++c) o[c] = f2bf(v[c]);
    *(bf16x4*)(out + (size_t)i * 4) = o;
  }
}

// in [K,N] f32 row-major -> out [N,K] bf16 row-major, via 32x32 LDS tile
__global__ __launch_bounds__(256) void cvt_transpose_bf16(const float* __restrict__ in,
                                                          __bf16* __restrict__ out,
                                                          int N, int K) {
  __shared__ float tile[32][33];
  const int tx = threadIdx.x;      // 0..31
  const int ty = threadIdx.y;      // 0..7
  const int n0 = blockIdx.x * 32;
  const int k0 = blockIdx.y * 32;
#pragma unroll
  for (int r = 0; r < 4; ++r)
    tile[ty + r * 8][tx] = in[(size_t)(k0 + ty + r * 8) * N + n0 + tx];
  __syncthreads();
#pragma unroll
  for (int r = 0; r < 4; ++r)
    out[(size_t)(n0 + ty + r * 8) * K + k0 + tx] = f2bf(tile[tx][ty + r * 8]);
}

// ---------------------------------------------------------------------------
// QKV GEMM: qkv[m, n] = sum_k x[m,k] * Wqkv[k,n] + b[n]
//   A = x_bf [8192,1024] row-major, B = WqkvT [3072,1024] (k contiguous)
//   wave -> 32x64 output tile (8 WMMA / 12 b128 per 32-k step);
//   k-loop unrolled x2 with register double-buffer
// ---------------------------------------------------------------------------
__global__ __launch_bounds__(128) void qkv_gemm(const __bf16* __restrict__ xb,
                                                const __bf16* __restrict__ wt,
                                                const float*  __restrict__ bias,
                                                __bf16* __restrict__ qb,
                                                __bf16* __restrict__ kb,
                                                __bf16* __restrict__ vtb) {
  const int wave = threadIdx.x >> 5;
  const int lane = threadIdx.x & 31;
  const int col  = lane & 15;
  const int hl   = lane >> 4;
  const int m0 = blockIdx.y * 128 + wave * 32;   // 32 rows per wave
  const int n0 = blockIdx.x * 64;

  const __bf16* arow[2];
  arow[0] = xb + (size_t)m0 * CC;
  arow[1] = xb + (size_t)(m0 + 16) * CC;
  const __bf16* wrow[4];
#pragma unroll
  for (int j = 0; j < 4; ++j) wrow[j] = wt + (size_t)(n0 + j * 16) * CC;

  v8f acc[2][4] = {};
  v16bf a_cur[2], b_cur[4];
#pragma unroll
  for (int i = 0; i < 2; ++i) a_cur[i] = load_frag(arow[i], CC, 0);
#pragma unroll
  for (int j = 0; j < 4; ++j) b_cur[j] = load_frag(wrow[j], CC, 0);

  for (int k0 = 0; k0 < CC; k0 += 64) {
    // stage n+1 loads issued before stage n math (overlap VMEM with WMMA)
    v16bf a_nxt[2], b_nxt[4];
#pragma unroll
    for (int i = 0; i < 2; ++i) a_nxt[i] = load_frag(arow[i], CC, k0 + 32);
#pragma unroll
    for (int j = 0; j < 4; ++j) b_nxt[j] = load_frag(wrow[j], CC, k0 + 32);
#pragma unroll
    for (int i = 0; i < 2; ++i) prefetch_frag(arow[i], CC, k0 + 128);
#pragma unroll
    for (int j = 0; j < 4; ++j) prefetch_frag(wrow[j], CC, k0 + 128);

#pragma unroll
    for (int j = 0; j < 4; ++j)
#pragma unroll
      for (int i = 0; i < 2; ++i)
        acc[i][j] = __builtin_amdgcn_wmma_f32_16x16x32_bf16(false, a_cur[i], false, b_cur[j],
                                                            (short)0, acc[i][j], false, false);
    const int k2 = k0 + 64;
    if (k2 < CC) {
#pragma unroll
      for (int i = 0; i < 2; ++i) a_cur[i] = load_frag(arow[i], CC, k2);
#pragma unroll
      for (int j = 0; j < 4; ++j) b_cur[j] = load_frag(wrow[j], CC, k2);
    }
#pragma unroll
    for (int j = 0; j < 4; ++j)
#pragma unroll
      for (int i = 0; i < 2; ++i)
        acc[i][j] = __builtin_amdgcn_wmma_f32_16x16x32_bf16(false, a_nxt[i], false, b_nxt[j],
                                                            (short)0, acc[i][j], false, false);
  }

#pragma unroll
  for (int j = 0; j < 4; ++j) {
    const int n = n0 + j * 16 + col;
    const float bn = bias[n];
#pragma unroll
    for (int i = 0; i < 2; ++i)
#pragma unroll
      for (int r = 0; r < 8; ++r) {
        const float val = acc[i][j][r] + bn;
        const int m = m0 + i * 16 + r + hl * 8;  // global row in [0, 8192)
        const int b = m >> 11, t = m & (TT - 1);
        if (n < CC) {                            // Q -> [B,H,T,D]
          const int h = n >> 6, d = n & 63;
          qb[(((size_t)(b * HH + h)) * TT + t) * DD + d] = f2bf(val);
        } else if (n < 2 * CC) {                 // K -> [B,H,T,D]
          const int n2 = n - CC, h = n2 >> 6, d = n2 & 63;
          kb[(((size_t)(b * HH + h)) * TT + t) * DD + d] = f2bf(val);
        } else {                                 // V -> [B,H,D,T] (transposed)
          const int n2 = n - 2 * CC, h = n2 >> 6, d = n2 & 63;
          vtb[(((size_t)(b * HH + h)) * DD + d) * TT + t] = f2bf(val);
        }
      }
  }
}

// ---------------------------------------------------------------------------
// Flash attention: one wave per 16-query tile, 32-key tiles, online softmax.
// grid = (T/64, B*H), block = 128 (4 waves; all 4 share K/V through WGP$)
// ---------------------------------------------------------------------------
__global__ __launch_bounds__(128) void attn_flash(const __bf16* __restrict__ qb,
                                                  const __bf16* __restrict__ kb,
                                                  const __bf16* __restrict__ vtb,
                                                  __bf16* __restrict__ ob) {
  __shared__ __align__(16) __bf16 plds[4][16 * 32];

  const int wave = threadIdx.x >> 5;
  const int lane = threadIdx.x & 31;
  const int col  = lane & 15;
  const int hl   = lane >> 4;
  const int bh = blockIdx.y;                    // b*16 + h
  const int q0 = blockIdx.x * 64 + wave * 16;   // query tile base within T

  const __bf16* qbase = qb  + (size_t)bh * TT * DD;
  const __bf16* kbase = kb  + (size_t)bh * TT * DD;
  const __bf16* vbase = vtb + (size_t)bh * DD * TT;

  const v16bf qa0 = load_frag(qbase + (size_t)q0 * DD, DD, 0);
  const v16bf qa1 = load_frag(qbase + (size_t)q0 * DD, DD, 32);

  float mi[8], li[8];
#pragma unroll
  for (int r = 0; r < 8; ++r) { mi[r] = -1e30f; li[r] = 0.0f; }
  v8f acc[4] = {};

  const int ktiles = (q0 + 16 + 31) >> 5;       // causal: keys <= q0+15
  for (int kt = 0; kt < ktiles; ++kt) {
    const int key0 = kt * 32;

    // ---- issue ALL loads first: K frags, then V frags.  Score WMMAs only
    // need K (in-order completion leaves V in flight under the softmax).
    v16bf kf[2][2];
#pragma unroll
    for (int nt = 0; nt < 2; ++nt) {
      const __bf16* krow = kbase + (size_t)(key0 + nt * 16) * DD;
      kf[nt][0] = load_frag(krow, DD, 0);
      kf[nt][1] = load_frag(krow, DD, 32);
    }
    v16bf vf[4];
#pragma unroll
    for (int dt = 0; dt < 4; ++dt)
      vf[dt] = load_frag(vbase + (size_t)(dt * 16) * TT, TT, key0);

    if (kt + 1 < ktiles) {                       // prefetch next key tile
      prefetch_frag(kbase + (size_t)(key0 + 32) * DD, DD, 0);
      prefetch_frag(kbase + (size_t)(key0 + 48) * DD, DD, 0);
#pragma unroll
      for (int dt = 0; dt < 4; ++dt)
        prefetch_frag(vbase + (size_t)(dt * 16) * TT, TT, key0 + 32);
    }

    // ---- scores S = (Q K^T) / sqrt(D), two 16-col halves -----------------
    v8f s[2];
#pragma unroll
    for (int nt = 0; nt < 2; ++nt) {
      v8f t = {};
      t = __builtin_amdgcn_wmma_f32_16x16x32_bf16(false, qa0, false, kf[nt][0], (short)0, t, false, false);
      t = __builtin_amdgcn_wmma_f32_16x16x32_bf16(false, qa1, false, kf[nt][1], (short)0, t, false, false);
      s[nt] = t;
    }

    // ---- scale + causal mask (C layout: row = r + hl*8, col = lane&15) ---
#pragma unroll
    for (int nt = 0; nt < 2; ++nt)
#pragma unroll
      for (int r = 0; r < 8; ++r) {
        const int key = key0 + nt * 16 + col;
        const int qr  = q0 + r + hl * 8;
        s[nt][r] = (key > qr) ? -1e30f : s[nt][r] * 0.125f;
      }

    // ---- online softmax (V loads still in flight here) --------------------
#pragma unroll
    for (int r = 0; r < 8; ++r) {
      float v = fmaxf(s[0][r], s[1][r]);
#pragma unroll
      for (int off = 1; off < 16; off <<= 1) v = fmaxf(v, __shfl_xor(v, off, 32));
      const float mnew  = fmaxf(mi[r], v);
      const float alpha = __expf(mi[r] - mnew);
      const float p0 = __expf(s[0][r] - mnew);
      const float p1 = __expf(s[1][r] - mnew);
      s[0][r] = p0; s[1][r] = p1;
      float rs = p0 + p1;
#pragma unroll
      for (int off = 1; off < 16; off <<= 1) rs += __shfl_xor(rs, off, 32);
      li[r] = li[r] * alpha + rs;
      mi[r] = mnew;
#pragma unroll
      for (int dt = 0; dt < 4; ++dt) acc[dt][r] *= alpha;
    }

    // ---- stage P (C layout) to LDS, reload as A fragment ------------------
#pragma unroll
    for (int nt = 0; nt < 2; ++nt)
#pragma unroll
      for (int r = 0; r < 8; ++r)
        plds[wave][(r + hl * 8) * 32 + nt * 16 + col] = f2bf(s[nt][r]);
    // per-wave LDS ops are in-order: ds_store then ds_load, no barrier needed
    v16bf pa;
    {
      const __bf16* pp = &plds[wave][col * 32 + hl * 8];
      bf16x8 lo = *(const bf16x8*)(pp);
      bf16x8 hi = *(const bf16x8*)(pp + 16);
#pragma unroll
      for (int i = 0; i < 8; ++i) { pa[i] = lo[i]; pa[i + 8] = hi[i]; }
    }

    // ---- O += P * V  (V fragments were loaded at the top of the tile) -----
#pragma unroll
    for (int dt = 0; dt < 4; ++dt)
      acc[dt] = __builtin_amdgcn_wmma_f32_16x16x32_bf16(false, pa, false, vf[dt],
                                                        (short)0, acc[dt], false, false);
  }

  // ---- normalize and write attn bf16 [B,T,C] -------------------------------
  const int b = bh >> 4, h = bh & 15;
#pragma unroll
  for (int r = 0; r < 8; ++r) {
    const float inv = 1.0f / li[r];
    const int t = q0 + r + hl * 8;
    const size_t rowoff = ((size_t)(b * TT + t)) * CC + h * DD;
#pragma unroll
    for (int dt = 0; dt < 4; ++dt)
      ob[rowoff + dt * 16 + col] = f2bf(acc[dt][r] * inv);
  }
}

// ---------------------------------------------------------------------------
// Output projection: out[m,n] = sum_k attn[m,k] * Wp[k,n] + bp[n]   (f32 out)
//   wave -> 32x64 tile, reg double-buffered
// ---------------------------------------------------------------------------
__global__ __launch_bounds__(128) void proj_gemm(const __bf16* __restrict__ ab,
                                                 const __bf16* __restrict__ wt,
                                                 const float*  __restrict__ bias,
                                                 float* __restrict__ out) {
  const int wave = threadIdx.x >> 5;
  const int lane = threadIdx.x & 31;
  const int col  = lane & 15;
  const int hl   = lane >> 4;
  const int m0 = blockIdx.y * 128 + wave * 32;
  const int n0 = blockIdx.x * 64;

  const __bf16* arow[2];
  arow[0] = ab + (size_t)m0 * CC;
  arow[1] = ab + (size_t)(m0 + 16) * CC;
  const __bf16* wrow[4];
#pragma unroll
  for (int j = 0; j < 4; ++j) wrow[j] = wt + (size_t)(n0 + j * 16) * CC;

  v8f acc[2][4] = {};
  v16bf a_cur[2], b_cur[4];
#pragma unroll
  for (int i = 0; i < 2; ++i) a_cur[i] = load_frag(arow[i], CC, 0);
#pragma unroll
  for (int j = 0; j < 4; ++j) b_cur[j] = load_frag(wrow[j], CC, 0);

  for (int k0 = 0; k0 < CC; k0 += 64) {
    v16bf a_nxt[2], b_nxt[4];
#pragma unroll
    for (int i = 0; i < 2; ++i) a_nxt[i] = load_frag(arow[i], CC, k0 + 32);
#pragma unroll
    for (int j = 0; j < 4; ++j) b_nxt[j] = load_frag(wrow[j], CC, k0 + 32);
#pragma unroll
    for (int i = 0; i < 2; ++i) prefetch_frag(arow[i], CC, k0 + 128);
#pragma unroll
    for (int j = 0; j < 4; ++j) prefetch_frag(wrow[j], CC, k0 + 128);

#pragma unroll
    for (int j = 0; j < 4; ++j)
#pragma unroll
      for (int i = 0; i < 2; ++i)
        acc[i][j] = __builtin_amdgcn_wmma_f32_16x16x32_bf16(false, a_cur[i], false, b_cur[j],
                                                            (short)0, acc[i][j], false, false);
    const int k2 = k0 + 64;
    if (k2 < CC) {
#pragma unroll
      for (int i = 0; i < 2; ++i) a_cur[i] = load_frag(arow[i], CC, k2);
#pragma unroll
      for (int j = 0; j < 4; ++j) b_cur[j] = load_frag(wrow[j], CC, k2);
    }
#pragma unroll
    for (int j = 0; j < 4; ++j)
#pragma unroll
      for (int i = 0; i < 2; ++i)
        acc[i][j] = __builtin_amdgcn_wmma_f32_16x16x32_bf16(false, a_nxt[i], false, b_nxt[j],
                                                            (short)0, acc[i][j], false, false);
  }
#pragma unroll
  for (int j = 0; j < 4; ++j) {
    const int n = n0 + j * 16 + col;
    const float bn = bias[n];
#pragma unroll
    for (int i = 0; i < 2; ++i)
#pragma unroll
      for (int r = 0; r < 8; ++r) {
        const int m = m0 + i * 16 + r + hl * 8;
        out[(size_t)m * CC + n] = acc[i][j][r] + bn;
      }
  }
}

// ---------------------------------------------------------------------------
extern "C" void kernel_launch(void* const* d_in, const int* in_sizes, int n_in,
                              void* d_out, int out_size, void* d_ws, size_t ws_size,
                              hipStream_t stream) {
  const float* x      = (const float*)d_in[0];   // [B,T,C]
  const float* W_qkv  = (const float*)d_in[1];   // [C,3C]
  const float* b_qkv  = (const float*)d_in[2];   // [3C]
  const float* W_proj = (const float*)d_in[3];   // [C,C]
  const float* b_proj = (const float*)d_in[4];   // [C]
  float* out = (float*)d_out;                    // [B,T,C]

  // workspace layout (bf16 buffers), total = 88 MiB
  char* ws = (char*)d_ws;
  __bf16* xb    = (__bf16*)(ws);                                  // 8192*1024
  __bf16* wqkvT = (__bf16*)(ws + 16777216);                       // 3072*1024
  __bf16* wpT   = (__bf16*)(ws + 16777216 + 6291456);             // 1024*1024
  __bf16* qb    = (__bf16*)(ws + 25165824);                       // [B,H,T,D]
  __bf16* kb    = (__bf16*)(ws + 41943040);                       // [B,H,T,D]
  __bf16* vtb   = (__bf16*)(ws + 58720256);                       // [B,H,D,T]
  __bf16* attnb = (__bf16*)(ws + 75497472);                       // [B,T,C]

  const int nX4 = (MROWS * CC) / 4;              // 2097152
  cvt_f32_bf16_v4<<<(nX4 + 255) / 256, 256, 0, stream>>>(x, xb, nX4);
  cvt_transpose_bf16<<<dim3(NQKV / 32, CC / 32), dim3(32, 8), 0, stream>>>(W_qkv, wqkvT, NQKV, CC);
  cvt_transpose_bf16<<<dim3(CC / 32, CC / 32), dim3(32, 8), 0, stream>>>(W_proj, wpT, CC, CC);

  qkv_gemm<<<dim3(NQKV / 64, MROWS / 128), 128, 0, stream>>>(xb, wqkvT, b_qkv, qb, kb, vtb);
  attn_flash<<<dim3(TT / 64, BB * HH), 128, 0, stream>>>(qb, kb, vtb, attnb);
  proj_gemm<<<dim3(CC / 64, MROWS / 128), 128, 0, stream>>>(attnb, wpT, b_proj, out);
}
